// CMDTLoss_18236431139468
// MI455X (gfx1250) — matrix operand order repaired and checked by reading, not compiled, not run on hardware
//
#include <hip/hip_runtime.h>
#include <hip/hip_bf16.h>

// CMDT supervised-contrastive loss for MI455X (gfx1250).
// Math reduction: FFT cancels by Plancherel -> cosine-sim GEMM of real features.
// GEMM runs on v_wmma_f32_16x16x32_bf16 (bf16x3 split for fp32 accuracy) with
// double-buffered GLOBAL_LOAD_ASYNC_TO_LDS_B128 staging (ASYNCcnt pipeline).

typedef __attribute__((ext_vector_type(16))) __bf16        v16bf;
typedef __attribute__((ext_vector_type(8)))  float         v8f;
typedef __attribute__((ext_vector_type(4)))  unsigned int  u32x4;
typedef __attribute__((ext_vector_type(8)))  unsigned int  u32x8;

#define N_ROWS 4096
#define D_K    512
#define TILE   128
#define KSTEP  32
#define JBLK   (N_ROWS / TILE)
#define LDSTR  40          // padded LDS row stride in bf16 elements (80 B)
#define INV_T  10.0f
#define EPSF   1e-8f

// ---------------------------------------------------------------- split f32 -> bf16 hi/lo
__global__ __launch_bounds__(256) void cmdt_split(const float* __restrict__ f,
                                                  unsigned short* __restrict__ fh,
                                                  unsigned short* __restrict__ fl,
                                                  int n) {
  for (int i = blockIdx.x * blockDim.x + threadIdx.x; i < n; i += gridDim.x * blockDim.x) {
    float v = f[i];
    unsigned int u = __float_as_uint(v);
    unsigned short h16 = (unsigned short)((u + 0x7FFFu + ((u >> 16) & 1u)) >> 16);
    float hf = __uint_as_float(((unsigned int)h16) << 16);
    float lo = v - hf;
    unsigned int ul = __float_as_uint(lo);
    unsigned short l16 = (unsigned short)((ul + 0x7FFFu + ((ul >> 16) & 1u)) >> 16);
    fh[i] = h16;
    fl[i] = l16;
  }
}

// ---------------------------------------------------------------- row inverse norms
__global__ __launch_bounds__(256) void cmdt_norm(const float* __restrict__ f,
                                                 float* __restrict__ invn) {
  int wave = threadIdx.x >> 5;
  int lane = threadIdx.x & 31;
  int row  = blockIdx.x * 8 + wave;
  const float* p = f + (size_t)row * D_K;
  float s = 0.f;
  #pragma unroll
  for (int t = 0; t < D_K / 32; ++t) {
    float v = p[lane + 32 * t];
    s += v * v;
  }
  #pragma unroll
  for (int off = 16; off >= 1; off >>= 1) s += __shfl_xor(s, off, 32);
  if (lane == 0) invn[row] = rsqrtf(s);
}

// ---------------------------------------------------------------- async copy helpers
// 32 bytes per lane: two GLOBAL_LOAD_ASYNC_TO_LDS_B128 in GVS mode
// (SGPR64 base + per-lane 32-bit byte offset). LDS dest address VGPR holds the
// wave-relative LDS byte offset (= low 32 bits of the generic shared pointer).
static __device__ inline void async_copy32(const unsigned short* gbase, unsigned voff,
                                           unsigned short* lptr) {
  unsigned l0 = (unsigned)(size_t)lptr;
  asm volatile("global_load_async_to_lds_b128 %0, %1, %2\n\t"
               "global_load_async_to_lds_b128 %3, %4, %2"
               :
               : "v"(l0), "v"(voff), "s"(gbase), "v"(l0 + 16u), "v"(voff + 16u)
               : "memory");
}

static __device__ inline void wait_async0() {
  asm volatile("s_wait_asynccnt 0" ::: "memory");
}

// ---------------------------------------------------------------- fragment helpers
static __device__ inline v16bf frag_pair(const unsigned short* p0, const unsigned short* p1) {
  u32x4 lo = *(const u32x4*)p0;
  u32x4 hi = *(const u32x4*)p1;
  u32x8 c;
  c[0] = lo[0]; c[1] = lo[1]; c[2] = lo[2]; c[3] = lo[3];
  c[4] = hi[0]; c[5] = hi[1]; c[6] = hi[2]; c[7] = hi[3];
  return __builtin_bit_cast(v16bf, c);
}

// A fragment, 16x32 bf16: lane M = lane%16, h = lane/16.
// elems 0..7  -> K = h*8 + e      (ushort off h*8)
// elems 8..15 -> K = 16 + h*8 + e (ushort off 16 + h*8)
static __device__ inline v16bf load_afrag(const unsigned short* sh, int rowBase, int nlane, int h) {
  const unsigned short* p = sh + (rowBase + nlane) * LDSTR + h * 8;
  return frag_pair(p, p + 16);
}

// B fragment, 32x16 bf16: lane N = lane%16, h = lane/16, elems 0..7 -> K = h*8 + e, elems 8..15 -> K = 16 + h*8 + e.
static __device__ inline v16bf load_bfrag(const unsigned short* sh, int colBase, int nlane, int h) {
  const unsigned short* p = sh + (colBase + nlane) * LDSTR + h * 8;
  return frag_pair(p, p + 16);
}

// ---------------------------------------------------------------- fused GEMM + epilogue
__global__ __launch_bounds__(256) void cmdt_gemm(const unsigned short* __restrict__ Fh,
                                                 const unsigned short* __restrict__ Fl,
                                                 const float* __restrict__ invn,
                                                 const long long* __restrict__ labels,
                                                 float* __restrict__ pE,
                                                 float* __restrict__ pS,
                                                 float* __restrict__ pP) {
  __shared__ unsigned short shAh[2][TILE * LDSTR];
  __shared__ unsigned short shAl[2][TILE * LDSTR];
  __shared__ unsigned short shBh[2][TILE * LDSTR];
  __shared__ unsigned short shBl[2][TILE * LDSTR];
  __shared__ float shInvI[TILE], shInvJ[TILE];
  __shared__ int   shLabI[TILE], shLabJ[TILE];
  __shared__ float shE[4][TILE], shS[4][TILE], shP[4][TILE];

  const int jb = blockIdx.x, ib = blockIdx.y;
  const int i0 = ib * TILE, j0 = jb * TILE;
  const int tid   = threadIdx.x;
  const int lane  = tid & 31;
  const int wave  = tid >> 5;
  const int wr    = wave & 1;    // 64-row group
  const int wc    = wave >> 1;   // 32-col group
  const int h     = lane >> 4;
  const int nlane = lane & 15;

  if (tid < TILE) {
    shInvI[tid] = invn[i0 + tid];
    shLabI[tid] = (int)labels[i0 + tid];
  } else {
    int t = tid - TILE;
    shInvJ[t] = invn[j0 + t];
    shLabJ[t] = (int)labels[j0 + t];
  }

  v8f zero = {0.f, 0.f, 0.f, 0.f, 0.f, 0.f, 0.f, 0.f};
  v8f acc[4][2];
  #pragma unroll
  for (int mi = 0; mi < 4; ++mi)
    #pragma unroll
    for (int nj = 0; nj < 2; ++nj) acc[mi][nj] = zero;

  const int rowT  = tid >> 1;    // 0..127
  const int halfT = tid & 1;     // which 16-element chunk of the 32-wide K slab
  const int soff  = rowT * LDSTR + halfT * 16;

  auto issue = [&](int buf, int k0) {
    unsigned voffA = (unsigned)((((i0 + rowT) * D_K) + k0 + halfT * 16) * 2);
    unsigned voffB = (unsigned)((((j0 + rowT) * D_K) + k0 + halfT * 16) * 2);
    async_copy32(Fh, voffA, &shAh[buf][soff]);
    async_copy32(Fl, voffA, &shAl[buf][soff]);
    async_copy32(Fh, voffB, &shBh[buf][soff]);
    async_copy32(Fl, voffB, &shBl[buf][soff]);
  };

  issue(0, 0);  // prologue: fill buffer 0

  for (int k0 = 0; k0 < D_K; k0 += KSTEP) {
    const int cur = (k0 >> 5) & 1;
    wait_async0();        // our async copies into `cur` (and any older) are done
    __syncthreads();      // everyone's copies are visible
    if (k0 + KSTEP < D_K) issue(cur ^ 1, k0 + KSTEP);  // overlap with compute

    const unsigned short* Ah = shAh[cur];
    const unsigned short* Al = shAl[cur];
    const unsigned short* Bh = shBh[cur];
    const unsigned short* Bl = shBl[cur];

    v16bf bh[2], bl[2];
    #pragma unroll
    for (int nj = 0; nj < 2; ++nj) {
      bh[nj] = load_bfrag(Bh, wc * 32 + nj * 16, nlane, h);
      bl[nj] = load_bfrag(Bl, wc * 32 + nj * 16, nlane, h);
    }
    #pragma unroll
    for (int mi = 0; mi < 4; ++mi) {
      v16bf ah = load_afrag(Ah, wr * 64 + mi * 16, nlane, h);
      v16bf al = load_afrag(Al, wr * 64 + mi * 16, nlane, h);
      #pragma unroll
      for (int nj = 0; nj < 2; ++nj) {
        // bf16x3: hi*hi + hi*lo + lo*hi  (lo*lo dropped, ~2^-18 relative)
        acc[mi][nj] = __builtin_amdgcn_wmma_f32_16x16x32_bf16(
            false, ah, false, bh[nj], (short)0, acc[mi][nj], false, false);
        acc[mi][nj] = __builtin_amdgcn_wmma_f32_16x16x32_bf16(
            false, ah, false, bl[nj], (short)0, acc[mi][nj], false, false);
        acc[mi][nj] = __builtin_amdgcn_wmma_f32_16x16x32_bf16(
            false, al, false, bh[nj], (short)0, acc[mi][nj], false, false);
      }
    }
    __syncthreads();      // all reads of `cur` done (syncthreads flushes DScnt)
  }

  // Epilogue: C/D layout -> lane N = lane%16, VGPR r holds M = r + 8*h.
  #pragma unroll
  for (int mi = 0; mi < 4; ++mi) {
    #pragma unroll
    for (int r = 0; r < 8; ++r) {
      int   li  = wr * 64 + mi * 16 + r + h * 8;
      int   gi  = i0 + li;
      float ri  = shInvI[li];
      int   lbi = shLabI[li];
      float e = 0.f, s = 0.f, p = 0.f;
      #pragma unroll
      for (int nj = 0; nj < 2; ++nj) {
        int   lj  = wc * 32 + nj * 16 + nlane;
        int   gj  = j0 + lj;
        float sim = acc[mi][nj][r] * ri * shInvJ[lj] * INV_T;
        bool  diag = (gi == gj);
        e += diag ? 0.f : expf(sim);
        bool  pos = (!diag) && (lbi == shLabJ[lj]);
        s += pos ? sim : 0.f;
        p += pos ? 1.f : 0.f;
      }
      // reduce across the 16 columns held by this lane group (fixed butterfly)
      #pragma unroll
      for (int off = 8; off >= 1; off >>= 1) {
        e += __shfl_xor(e, off, 32);
        s += __shfl_xor(s, off, 32);
        p += __shfl_xor(p, off, 32);
      }
      if (nlane == 0) {
        shE[wc][li] = e;
        shS[wc][li] = s;
        shP[wc][li] = p;
      }
    }
  }
  __syncthreads();

  if (tid < TILE) {
    float e = shE[0][tid] + shE[1][tid] + shE[2][tid] + shE[3][tid];
    float s = shS[0][tid] + shS[1][tid] + shS[2][tid] + shS[3][tid];
    float p = shP[0][tid] + shP[1][tid] + shP[2][tid] + shP[3][tid];
    size_t o = (size_t)jb * N_ROWS + (size_t)(i0 + tid);
    pE[o] = e;
    pS[o] = s;
    pP[o] = p;
  }
}

// ---------------------------------------------------------------- final deterministic reduce
__global__ __launch_bounds__(256) void cmdt_finalize(const float* __restrict__ pE,
                                                     const float* __restrict__ pS,
                                                     const float* __restrict__ pP,
                                                     float* __restrict__ out) {
  __shared__ float red[256];
  float local = 0.f;
  for (int i = threadIdx.x; i < N_ROWS; i += 256) {
    float e = 0.f, s = 0.f, p = 0.f;
    for (int jb = 0; jb < JBLK; ++jb) {
      size_t o = (size_t)jb * N_ROWS + (size_t)i;
      e += pE[o];
      s += pS[o];
      p += pP[o];
    }
    local += (s - p * logf(e)) / (p + EPSF);
  }
  red[threadIdx.x] = local;
  __syncthreads();
  for (int off = 128; off >= 1; off >>= 1) {
    if ((int)threadIdx.x < off) red[threadIdx.x] += red[threadIdx.x + off];
    __syncthreads();
  }
  if (threadIdx.x == 0) out[0] = -red[0] / (float)N_ROWS;
}

// ---------------------------------------------------------------- host entry
extern "C" void kernel_launch(void* const* d_in, const int* in_sizes, int n_in,
                              void* d_out, int out_size, void* d_ws, size_t ws_size,
                              hipStream_t stream) {
  (void)in_sizes; (void)n_in; (void)out_size; (void)ws_size;
  const float*     feats  = (const float*)d_in[0];
  const long long* labels = (const long long*)d_in[1];

  char* ws = (char*)d_ws;
  unsigned short* Fh   = (unsigned short*)(ws);                       // 4 MB
  unsigned short* Fl   = (unsigned short*)(ws + (size_t)4194304);     // 4 MB
  float*          invn = (float*)(ws + (size_t)8388608);              // 16 KB
  float*          pE   = (float*)(ws + (size_t)8404992);              // 512 KB
  float*          pS   = (float*)(ws + (size_t)8929280);              // 512 KB
  float*          pP   = (float*)(ws + (size_t)9453568);              // 512 KB
  float*          out  = (float*)d_out;

  cmdt_split<<<2048, 256, 0, stream>>>(feats, Fh, Fl, N_ROWS * D_K);
  cmdt_norm<<<N_ROWS / 8, 256, 0, stream>>>(feats, invn);
  dim3 grid(JBLK, N_ROWS / TILE);
  cmdt_gemm<<<grid, 256, 0, stream>>>(Fh, Fl, invn, labels, pE, pS, pP);
  cmdt_finalize<<<1, 256, 0, stream>>>(pE, pS, pP, out);
}
